// SpatialAttention_74929999446751
// MI455X (gfx1250) — compile-verified, matrix operand order checked by reading, not run on hardware
//
#include <hip/hip_runtime.h>
#include <hip/hip_bf16.h>

// SpatialAttention on MI455X (gfx1250), wave32 + bf16 WMMA + async LDS->HBM DMA.
//
// Per (n,t): S = K^T Q / 16  (1024x1024, contract C=256)
//            attn = softmax over KEY axis (columns of S)
//            O = V @ attn    (256x1024, contract P=1024)
//
// One workgroup = one (nt, 32-wide q-tile). LDS (of 320KB/WGP):
//   Sbuf     f32  [1024 x 32]        128 KB  scores -> exp -> normalized attn
//   AttnFrag bf16 (WMMA-B swizzled)   64 KB
//   QFrag    bf16 (WMMA-B swizzled)   16 KB
//   reduction scratch                ~2.3 KB
// After softmax, the normalized f32 panel is streamed to HBM with
// GLOBAL_STORE_ASYNC_FROM_LDS_B128 (ASYNCcnt DMA), fully overlapped with the
// GEMM2 WMMA work. Each B fragment (contiguous 32B/lane ds_load_b128 pair)
// now feeds 2 WMMAs (two output strips per wave).

typedef __attribute__((ext_vector_type(16))) __bf16 v16bf;
typedef __attribute__((ext_vector_type(8)))  float  v8f;

#define NTB   32
#define CDIM  256
#define PDIM  1024
#define QDIM  1024
#define QTILE 32
#define C2DIM 256

#define SBUF_BYTES  (PDIM * QTILE * 4)                  // 131072
#define AFRAG_OFF   SBUF_BYTES                          // 131072
#define AFRAG_BYTES ((PDIM / 32) * 2 * 32 * 16 * 2)     // 65536
#define QFRAG_OFF   (AFRAG_OFF + AFRAG_BYTES)           // 196608
#define QFRAG_BYTES ((CDIM / 32) * 2 * 32 * 16 * 2)     // 16384
#define RED_OFF     (QFRAG_OFF + QFRAG_BYTES)           // 212992
#define LDS_BYTES   (RED_OFF + (256 + 256 + 32 + 32) * 4)

// 16-bit A/B fragment layout (ISA 7.12.2): half h -> K index within a 32-chunk.
__device__ __forceinline__ int kmap16(int lane, int h) {
    int v = h >> 1;
    int e = h & 1;
    int g = (lane >> 4) & 1;
    return ((v & 4) << 2) | (g << 3) | ((v & 3) << 1) | e;
}
// Inverse: (k-within-32, column n) -> (lane, half) slot.
__device__ __forceinline__ void frag_slot(int kk, int n, int& lane, int& h) {
    int g = (kk >> 3) & 1;
    int e = kk & 1;
    int v = ((kk >> 4) & 1) * 4 + ((kk >> 1) & 3);
    lane = g * 16 + n;
    h = v * 2 + e;
}

__device__ __forceinline__ v8f wmma_bf16(v16bf a, v16bf b, v8f c) {
    return __builtin_amdgcn_wmma_f32_16x16x32_bf16(false, a, false, b,
                                                   (short)0, c, false, false);
}

__global__ __launch_bounds__(256)
void spatial_attn_fused(const float* __restrict__ key,
                        const float* __restrict__ query,
                        const float* __restrict__ value,
                        float* __restrict__ out_val,
                        float* __restrict__ out_attn) {
    extern __shared__ char smem[];
    float*  Sbuf     = (float*)smem;                 // [p][qq]
    __bf16* AttnFrag = (__bf16*)(smem + AFRAG_OFF);  // WMMA-B swizzled attn
    __bf16* QFrag    = (__bf16*)(smem + QFRAG_OFF);  // WMMA-B swizzled Q tile
    float*  pmax     = (float*)(smem + RED_OFF);     // [8][32]
    float*  psum     = pmax + 256;                   // [8][32]
    float*  cmax     = psum + 256;                   // [32]
    float*  cinv     = cmax + 32;                    // [32]

    const int nt   = blockIdx.x;
    const int q0g  = blockIdx.y * QTILE;
    const int tid  = threadIdx.x;
    const int lane = tid & 31;
    const int wave = tid >> 5;          // 8 waves

    const float* Kb = key   + (size_t)nt * CDIM  * PDIM;
    const float* Qb = query + (size_t)nt * CDIM  * QDIM;
    const float* Vb = value + (size_t)nt * C2DIM * PDIM;
    float* Ab = out_attn + (size_t)nt * PDIM  * QDIM;
    float* Ob = out_val  + (size_t)nt * C2DIM * QDIM;

    const int n  = lane & 15;           // B/C/D column within 16-wide subtile
    const int mb = (lane >> 4) << 3;    // C/D row base: 0 or 8
    const int g8 = (lane >> 4) << 3;    // fragment K offset for this half-wave

    // ---- Stage Q tile, swizzled to WMMA-B fragment layout ----
    for (int idx = tid; idx < CDIM * QTILE; idx += 256) {
        int c = idx >> 5, qq = idx & 31;                 // coalesced read over qq
        float val = Qb[(size_t)c * QDIM + q0g + qq];
        int l2, h2;
        frag_slot(c & 31, qq & 15, l2, h2);
        QFrag[((((c >> 5) * 2) + (qq >> 4)) * 32 + l2) * 16 + h2] = (__bf16)val;
    }
    __syncthreads();

    // ---- Phase 1: S = (K^T Q)/16 into Sbuf; 2 p-strips share each B frag ----
    for (int r = 0; r < 4; ++r) {
        const int p0a = (r * 16 + wave) * 16;
        const int p0b = (r * 16 + wave + 8) * 16;
        v8f acc[2][2] = {};
        for (int cb = 0; cb < CDIM / 32; ++cb) {
            const int c0 = cb * 32;
            v16bf a0, a1;               // A[m=p][k=c] = K[c][p], coalesced lanes
            #pragma unroll
            for (int h = 0; h < 16; ++h) {
                int k = kmap16(lane, h);
                const float* col = Kb + (size_t)(c0 + k) * PDIM + (lane & 15);
                a0[h] = (__bf16)col[p0a];
                a1[h] = (__bf16)col[p0b];
            }
            #pragma unroll
            for (int qt = 0; qt < 2; ++qt) {
                v16bf b = *(const v16bf*)&QFrag[((cb * 2 + qt) * 32 + lane) * 16];
                acc[0][qt] = wmma_bf16(a0, b, acc[0][qt]);
                acc[1][qt] = wmma_bf16(a1, b, acc[1][qt]);
            }
        }
        #pragma unroll
        for (int s = 0; s < 2; ++s) {
            int p0 = s ? p0b : p0a;
            #pragma unroll
            for (int qt = 0; qt < 2; ++qt)
                #pragma unroll
                for (int rr = 0; rr < 8; ++rr)
                    Sbuf[(p0 + mb + rr) * QTILE + qt * 16 + n] =
                        acc[s][qt][rr] * 0.0625f;
        }
    }
    __syncthreads();

    // ---- Phase 2: softmax over KEY axis, all 8 waves, bank-conflict-free ----
    {
        int qq = tid & 31, part = tid >> 5;
        int pbeg = part * (PDIM / 8), pend = pbeg + (PDIM / 8);
        float mx = -3.402823466e+38f;
        for (int p = pbeg; p < pend; ++p) mx = fmaxf(mx, Sbuf[p * QTILE + qq]);
        pmax[part * 32 + qq] = mx;
        __syncthreads();
        if (tid < 32) {
            float m = pmax[tid];
            #pragma unroll
            for (int j = 1; j < 8; ++j) m = fmaxf(m, pmax[j * 32 + tid]);
            cmax[tid] = m;
        }
        __syncthreads();
        float cm = cmax[qq];
        float s = 0.f;
        for (int p = pbeg; p < pend; ++p) {
            float e = __expf(Sbuf[p * QTILE + qq] - cm);
            Sbuf[p * QTILE + qq] = e;
            s += e;
        }
        psum[part * 32 + qq] = s;
        __syncthreads();
        if (tid < 32) {
            float t = 0.f;
            #pragma unroll
            for (int j = 0; j < 8; ++j) t += psum[j * 32 + tid];
            cinv[tid] = 1.0f / t;
        }
        __syncthreads();
    }

    // ---- Normalize in place (f32) + pack bf16 WMMA-B frags (paired p rows) ----
    for (int idx = tid; idx < (PDIM / 2) * QTILE; idx += 256) {
        int pp = idx >> 5, qq = idx & 31;
        int p = pp * 2;
        float inv = cinv[qq];
        float v0 = Sbuf[p * QTILE + qq] * inv;
        float v1 = Sbuf[(p + 1) * QTILE + qq] * inv;
        Sbuf[p * QTILE + qq]       = v0;
        Sbuf[(p + 1) * QTILE + qq] = v1;
        int l2, h2;
        frag_slot(p & 31, qq & 15, l2, h2);          // h2 even -> packed b32 store
        union { __bf16 b[2]; unsigned int u; } pk;
        pk.b[0] = (__bf16)v0;
        pk.b[1] = (__bf16)v1;
        *(unsigned int*)&AttnFrag[((((p >> 5) * 2) + (qq >> 4)) * 32 + l2) * 16 + h2] = pk.u;
    }
    __syncthreads();

    // ---- Async DMA: normalized attn panel LDS -> HBM, overlapped with GEMM2 ----
    // 16B per lane per instruction; 8192 chunks / 256 threads = 32 instrs each.
    {
        const float* AbBase = Ab + q0g;                       // uniform
        unsigned int lds0 = (unsigned int)(size_t)Sbuf;       // LDS byte offset
        for (int ch = tid; ch < (PDIM * QTILE) / 4; ch += 256) {
            int e0 = ch * 4;
            int p = e0 >> 5, q4 = e0 & 31;                    // 4 consecutive qq
            unsigned int ldsa = lds0 + (unsigned int)e0 * 4u;
            unsigned int goff = ((unsigned int)p * QDIM + (unsigned int)q4) * 4u;
            asm volatile("global_store_async_from_lds_b128 %0, %1, %2"
                         :: "v"(goff), "v"(ldsa), "s"(AbBase)
                         : "memory");
        }
    }

    // ---- Phase 3: O = V @ attn; 2 c2-strips share each B frag ----
    {
        const int c20a = wave * 16;
        const int c20b = (wave + 8) * 16;
        v8f acc[2][2] = {};
        const float* vrow0 = Vb + (size_t)(c20a + (lane & 15)) * PDIM;
        const float* vrow1 = Vb + (size_t)(c20b + (lane & 15)) * PDIM;
        for (int pb = 0; pb < PDIM / 32; ++pb) {
            const int p0 = pb * 32;
            v16bf a0, a1;               // A[m=c2][k=p] = V[c2][p], float2 loads
            #pragma unroll
            for (int v = 0; v < 8; ++v) {
                int kb = ((v & 4) << 2) | g8 | ((v & 3) << 1);
                float2 t0 = *(const float2*)(vrow0 + p0 + kb);
                float2 t1 = *(const float2*)(vrow1 + p0 + kb);
                a0[2 * v] = (__bf16)t0.x; a0[2 * v + 1] = (__bf16)t0.y;
                a1[2 * v] = (__bf16)t1.x; a1[2 * v + 1] = (__bf16)t1.y;
            }
            #pragma unroll
            for (int qt = 0; qt < 2; ++qt) {
                v16bf b = *(const v16bf*)&AttnFrag[((pb * 2 + qt) * 32 + lane) * 16];
                acc[0][qt] = wmma_bf16(a0, b, acc[0][qt]);
                acc[1][qt] = wmma_bf16(a1, b, acc[1][qt]);
            }
        }
        #pragma unroll
        for (int s = 0; s < 2; ++s) {
            int c20 = s ? c20b : c20a;
            #pragma unroll
            for (int qt = 0; qt < 2; ++qt)
                #pragma unroll
                for (int rr = 0; rr < 8; ++rr)
                    Ob[(size_t)(c20 + mb + rr) * QDIM + q0g + qt * 16 + n] =
                        acc[s][qt][rr];
        }
    }

    // Drain the attn DMA (s_endpgm would also wait, this makes it explicit).
    asm volatile("s_wait_asynccnt 0x0" ::: "memory");
}

extern "C" void kernel_launch(void* const* d_in, const int* in_sizes, int n_in,
                              void* d_out, int out_size, void* d_ws, size_t ws_size,
                              hipStream_t stream) {
    (void)in_sizes; (void)n_in; (void)out_size; (void)d_ws; (void)ws_size;
    const float* key   = (const float*)d_in[0];
    const float* query = (const float*)d_in[1];
    const float* value = (const float*)d_in[2];

    float* out_val  = (float*)d_out;                          // (4,8,256,32,32)
    float* out_attn = out_val + (size_t)NTB * C2DIM * QDIM;   // (4,8,32,32,32,32)

    (void)hipFuncSetAttribute((const void*)spatial_attn_fused,
                              hipFuncAttributeMaxDynamicSharedMemorySize,
                              (int)LDS_BYTES);

    dim3 grid(NTB, QDIM / QTILE, 1);   // 32 x 32 workgroups
    dim3 block(256, 1, 1);             // 8 wave32 waves
    spatial_attn_fused<<<grid, block, LDS_BYTES, stream>>>(key, query, value,
                                                           out_val, out_attn);
}
